// ImprovedPredictor_15204184227949
// MI455X (gfx1250) — compile-verified, matrix operand order checked by reading, not run on hardware
//
#include <hip/hip_runtime.h>

typedef __attribute__((ext_vector_type(16))) _Float16 v16h;
typedef __attribute__((ext_vector_type(8)))  _Float16 h8;
typedef __attribute__((ext_vector_type(8)))  float    v8f;
typedef __attribute__((ext_vector_type(4)))  float    f4;

#define WAVES 8
#define THREADS 256
constexpr int Dd  = 128;
constexpr int Hh  = 64;
constexpr int Qq  = 32;
constexpr int HOR = 28;

union V16 { v16h v; h8 h[2]; _Float16 e[16]; };

// ---- LDS (≈100.5 KB / workgroup; CDNA5 WGP has 320 KB) ----
__shared__ __align__(32) _Float16 s_whh[12 * 2 * 512];   // W_hh^T B-frags: 12 n-tiles x 2 k-chunks
__shared__ __align__(32) _Float16 s_wp [ 4 * 4 * 512];   // Wp^T   B-frags: 4 x 4
__shared__ __align__(32) _Float16 s_wg1[ 2 * 4 * 512];   // Wg1^T  B-frags: 2 x 4
__shared__ __align__(32) _Float16 s_wo1[ 2 * 2 * 512];   // Wo1^T  B-frags: 2 x 2
__shared__ __align__(32) _Float16 s_wg2[ 2 * 1 * 512];   // Wg2^T  B-frags: 2 x 1 (cols 28..31 zero)
__shared__ __align__(32) _Float16 s_hbuf[WAVES][16 * 64];// per-wave h bounce (f16, row-major)
__shared__ __align__(16) float    s_obuf[WAVES][16 * 32];// per-wave out-proj hidden
__shared__            float       s_xbuf[WAVES][16];     // per-wave autoregressive x
__shared__            float       s_pred[WAVES][16][HOR];// per-wave GRU predictions

__device__ __forceinline__ void lds_wait() {
  asm volatile("s_wait_dscnt 0" ::: "memory");
}

__device__ __forceinline__ float sigm(float x) {
  return __builtin_amdgcn_rcpf(1.0f + __expf(-x));
}
__device__ __forceinline__ float tanh_fast(float x) {
  float e = __expf(2.0f * x);
  return (e - 1.0f) * __builtin_amdgcn_rcpf(e + 1.0f);
}

// Stage W (row-major [Nout, K]) as B-operand fragments of W^T for
// V_WMMA_F32_16X16X32_F16. Frag layout: dst[f*512 + lane*16 + j], where
// f = nt*nKC + kc, column n = nt*16 + (lane&15), khalf = lane>>4,
// k = kc*32 + (j<8 ? khalf*8 + j : 16 + khalf*8 + (j-8)).
__device__ __forceinline__ void stage_bfrags(_Float16* dst, const float* __restrict__ W,
                                             int Nout, int K, int nNT, int nKC, int tid) {
  int total = nNT * nKC * 512;
  for (int idx = tid; idx < total; idx += THREADS) {
    int f      = idx >> 9;
    int within = idx & 511;
    int lane   = within >> 4;
    int j      = within & 15;
    int kc     = f % nKC;
    int nt     = f / nKC;
    int n      = nt * 16 + (lane & 15);
    int khalf  = lane >> 4;
    int k      = kc * 32 + (j < 8 ? khalf * 8 + j : 16 + khalf * 8 + (j - 8));
    float v    = (n < Nout) ? W[(size_t)n * K + k] : 0.0f;
    dst[idx]   = (_Float16)v;
  }
}

__device__ __forceinline__ v16h bfrag(const _Float16* buf, int f, int lane) {
  return *(const v16h*)(buf + f * 512 + lane * 16);
}

// A fragment from an LDS row (row-major). p points at row start + kc*32.
__device__ __forceinline__ v16h afrag_lds(const _Float16* p, int khalf) {
  V16 u;
  u.h[0] = *(const h8*)(p + khalf * 8);
  u.h[1] = *(const h8*)(p + khalf * 8 + 16);
  return u.v;
}

// A fragment from a global f32 row. p points at row start + kc*32 + khalf*8.
__device__ __forceinline__ v16h afrag_gmem(const float* __restrict__ p) {
  f4 a = *(const f4*)(p);
  f4 b = *(const f4*)(p + 4);
  f4 c = *(const f4*)(p + 16);
  f4 d = *(const f4*)(p + 20);
  V16 u;
#pragma unroll
  for (int i = 0; i < 4; i++) {
    u.e[i]      = (_Float16)a[i];
    u.e[4 + i]  = (_Float16)b[i];
    u.e[8 + i]  = (_Float16)c[i];
    u.e[12 + i] = (_Float16)d[i];
  }
  return u.v;
}

#define WMMA(Afr, Bfr, Cacc) \
  __builtin_amdgcn_wmma_f32_16x16x32_f16(false, (Afr), false, (Bfr), (short)0, (Cacc), false, false)

__global__ __launch_bounds__(THREADS)
void improved_predictor_kernel(const float* __restrict__ features,
                               const float* __restrict__ last_value,
                               const float* __restrict__ W_ih,
                               const float* __restrict__ W_hh,
                               const float* __restrict__ b_ih,
                               const float* __restrict__ b_hh,
                               const float* __restrict__ Wp,
                               const float* __restrict__ bp,
                               const float* __restrict__ Wo1,
                               const float* __restrict__ bo1,
                               const float* __restrict__ Wo2,
                               const float* __restrict__ bo2,
                               const float* __restrict__ Wg1,
                               const float* __restrict__ bg1,
                               const float* __restrict__ Wg2,
                               const float* __restrict__ bg2,
                               const float* __restrict__ log_decay,
                               float* __restrict__ out) {
  const int tid = threadIdx.x;

  // ---- one-time weight staging into WMMA fragment layout (f16) ----
  stage_bfrags(s_whh, W_hh, 192, 64, 12, 2, tid);
  stage_bfrags(s_wp,  Wp,    64, 128, 4, 4, tid);
  stage_bfrags(s_wg1, Wg1,   32, 128, 2, 4, tid);
  stage_bfrags(s_wo1, Wo1,   32, 64,  2, 2, tid);
  stage_bfrags(s_wg2, Wg2,   28, 32,  2, 1, tid);
  __syncthreads();

  const int wave  = tid >> 5;
  const int lane  = tid & 31;
  const int l15   = lane & 15;
  const int khalf = lane >> 4;
  const size_t rowbase = ((size_t)blockIdx.x * WAVES + wave) * 16;

  // ---- per-lane uniform-ish scalars (loaded once) ----
  float wih_s[12], bih_s[12], bhh_s[12];
#pragma unroll
  for (int c = 0; c < 12; c++) {
    int col  = c * 16 + l15;
    wih_s[c] = W_ih[col];
    bih_s[c] = b_ih[col];
    bhh_s[c] = b_hh[col];
  }
  float bp_s[4];
#pragma unroll
  for (int nt = 0; nt < 4; nt++) bp_s[nt] = bp[nt * 16 + l15];
  float bo1_s[2], bg1_s[2], bg2_s[2];
#pragma unroll
  for (int nt = 0; nt < 2; nt++) {
    bo1_s[nt] = bo1[nt * 16 + l15];
    bg1_s[nt] = bg1[nt * 16 + l15];
    int col   = nt * 16 + l15;
    bg2_s[nt] = (col < HOR) ? bg2[col] : 0.0f;
  }
  const float bo2v = bo2[0];
  const float ldec = __expf(log_decay[0]);

  // last_value: for decay blend (C-layout rows) + initial x
  float lv[8];
#pragma unroll
  for (int v = 0; v < 8; v++) lv[v] = last_value[rowbase + v + 8 * khalf];
  if (lane < 16) s_xbuf[wave][lane] = last_value[rowbase + lane];

  // ---- hidden_proj + refine_gate layer 1 (share feature A-fragments) ----
  v8f hacc[4] = {};
  v8f g1acc[2] = {};
  const float* frow = features + (rowbase + l15) * Dd;
#pragma unroll
  for (int kc = 0; kc < 4; kc++) {
    v16h a = afrag_gmem(frow + kc * 32 + khalf * 8);
#pragma unroll
    for (int nt = 0; nt < 4; nt++) hacc[nt] = WMMA(a, bfrag(s_wp, nt * 4 + kc, lane), hacc[nt]);
#pragma unroll
    for (int nt = 0; nt < 2; nt++) g1acc[nt] = WMMA(a, bfrag(s_wg1, nt * 4 + kc, lane), g1acc[nt]);
  }
  float h[4][8];
#pragma unroll
  for (int nt = 0; nt < 4; nt++)
#pragma unroll
    for (int v = 0; v < 8; v++) h[nt][v] = hacc[nt][v] + bp_s[nt];

  // g1 = relu(...); bounce through hbuf ([16][32] f16) to transpose into A layout
#pragma unroll
  for (int nt = 0; nt < 2; nt++)
#pragma unroll
    for (int v = 0; v < 8; v++) {
      float g = fmaxf(g1acc[nt][v] + bg1_s[nt], 0.0f);
      s_hbuf[wave][(v + 8 * khalf) * 32 + nt * 16 + l15] = (_Float16)g;
    }
  lds_wait();

  // ---- refine_gate layer 2 + sigmoid (kept in registers, C layout) ----
  v8f gacc[2] = {};
  {
    v16h ag = afrag_lds(s_hbuf[wave] + l15 * 32, khalf);
#pragma unroll
    for (int nt = 0; nt < 2; nt++) gacc[nt] = WMMA(ag, bfrag(s_wg2, nt, lane), gacc[nt]);
  }
  float gate[2][8];
#pragma unroll
  for (int nt = 0; nt < 2; nt++)
#pragma unroll
    for (int v = 0; v < 8; v++) gate[nt][v] = sigm(gacc[nt][v] + bg2_s[nt]);

  // ---- store h0 to hbuf ([16][64] f16) for step 0 ----
#pragma unroll
  for (int nt = 0; nt < 4; nt++)
#pragma unroll
    for (int v = 0; v < 8; v++)
      s_hbuf[wave][(v + 8 * khalf) * 64 + nt * 16 + l15] = (_Float16)h[nt][v];
  lds_wait();

  // ============================ GRU recurrence ============================
#pragma unroll 1
  for (int t = 0; t < HOR; t++) {
    float xv[8];
#pragma unroll
    for (int v = 0; v < 8; v++) xv[v] = s_xbuf[wave][v + 8 * khalf];

    const _Float16* hrow = s_hbuf[wave] + l15 * 64;
    v16h a0 = afrag_lds(hrow, khalf);        // K 0..31
    v16h a1 = afrag_lds(hrow + 32, khalf);   // K 32..63

    // r gate (frags 0..3)
    float r[4][8];
#pragma unroll
    for (int nt = 0; nt < 4; nt++) {
      v8f acc = {};
      acc = WMMA(a0, bfrag(s_whh, nt * 2 + 0, lane), acc);
      acc = WMMA(a1, bfrag(s_whh, nt * 2 + 1, lane), acc);
#pragma unroll
      for (int v = 0; v < 8; v++)
        r[nt][v] = sigm(acc[v] + xv[v] * wih_s[nt] + bih_s[nt] + bhh_s[nt]);
    }
    // n candidate (frags 8..11): tanh(gi_n + r * gh_n)
    float nc[4][8];
#pragma unroll
    for (int nt = 0; nt < 4; nt++) {
      v8f acc = {};
      acc = WMMA(a0, bfrag(s_whh, (8 + nt) * 2 + 0, lane), acc);
      acc = WMMA(a1, bfrag(s_whh, (8 + nt) * 2 + 1, lane), acc);
#pragma unroll
      for (int v = 0; v < 8; v++)
        nc[nt][v] = tanh_fast(xv[v] * wih_s[8 + nt] + bih_s[8 + nt] +
                              r[nt][v] * (acc[v] + bhh_s[8 + nt]));
    }
    // z gate (frags 4..7) + state update
#pragma unroll
    for (int nt = 0; nt < 4; nt++) {
      v8f acc = {};
      acc = WMMA(a0, bfrag(s_whh, (4 + nt) * 2 + 0, lane), acc);
      acc = WMMA(a1, bfrag(s_whh, (4 + nt) * 2 + 1, lane), acc);
#pragma unroll
      for (int v = 0; v < 8; v++) {
        float z = sigm(acc[v] + xv[v] * wih_s[4 + nt] + bih_s[4 + nt] + bhh_s[4 + nt]);
        h[nt][v] = (1.0f - z) * nc[nt][v] + z * h[nt][v];
      }
    }

    // store h_new -> hbuf (also serves as next step's input)
#pragma unroll
    for (int nt = 0; nt < 4; nt++)
#pragma unroll
      for (int v = 0; v < 8; v++)
        s_hbuf[wave][(v + 8 * khalf) * 64 + nt * 16 + l15] = (_Float16)h[nt][v];
    lds_wait();

    // output projection: relu(h_new @ Wo1^T + bo1)
    v16h oa0 = afrag_lds(hrow, khalf);
    v16h oa1 = afrag_lds(hrow + 32, khalf);
    v8f oacc[2] = {};
#pragma unroll
    for (int nt = 0; nt < 2; nt++) {
      oacc[nt] = WMMA(oa0, bfrag(s_wo1, nt * 2 + 0, lane), oacc[nt]);
      oacc[nt] = WMMA(oa1, bfrag(s_wo1, nt * 2 + 1, lane), oacc[nt]);
    }
#pragma unroll
    for (int nt = 0; nt < 2; nt++)
#pragma unroll
      for (int v = 0; v < 8; v++)
        s_obuf[wave][(v + 8 * khalf) * 32 + nt * 16 + l15] =
            fmaxf(oacc[nt][v] + bo1_s[nt], 0.0f);
    lds_wait();

    // Q->1 projection (tiny): lanes 0..15, one row each
    if (lane < 16) {
      float acc = bo2v;
      const f4* orow = (const f4*)(s_obuf[wave] + lane * 32);
#pragma unroll
      for (int q = 0; q < 8; q++) {
        f4 o = orow[q];
        f4 w = *(const f4*)(Wo2 + q * 4);
        acc += o[0] * w[0] + o[1] * w[1] + o[2] * w[2] + o[3] * w[3];
      }
      s_xbuf[wave][lane]    = acc;   // autoregressive input for next step
      s_pred[wave][lane][t] = acc;
    }
    lds_wait();
  }

  // ---- final blend: gate*gru_pred + (1-gate)*last_value*exp(-exp(ld)*t) ----
#pragma unroll
  for (int ct = 0; ct < 2; ct++) {
    int hc = ct * 16 + l15;
    if (hc < HOR) {
      float dc = __expf(-ldec * (float)(hc + 1));
#pragma unroll
      for (int v = 0; v < 8; v++) {
        int m    = v + 8 * khalf;
        float pr = s_pred[wave][m][hc];
        float g  = gate[ct][v];
        out[(rowbase + m) * HOR + hc] = g * pr + (1.0f - g) * dc * lv[v];
      }
    }
  }
}

extern "C" void kernel_launch(void* const* d_in, const int* in_sizes, int n_in,
                              void* d_out, int out_size, void* d_ws, size_t ws_size,
                              hipStream_t stream) {
  const float* features   = (const float*)d_in[0];
  const float* last_value = (const float*)d_in[1];
  const float* W_ih       = (const float*)d_in[2];
  const float* W_hh       = (const float*)d_in[3];
  const float* b_ih       = (const float*)d_in[4];
  const float* b_hh       = (const float*)d_in[5];
  const float* Wp         = (const float*)d_in[6];
  const float* bp         = (const float*)d_in[7];
  const float* Wo1        = (const float*)d_in[8];
  const float* bo1        = (const float*)d_in[9];
  const float* Wo2        = (const float*)d_in[10];
  const float* bo2        = (const float*)d_in[11];
  const float* Wg1        = (const float*)d_in[12];
  const float* bg1        = (const float*)d_in[13];
  const float* Wg2        = (const float*)d_in[14];
  const float* bg2        = (const float*)d_in[15];
  const float* log_decay  = (const float*)d_in[16];
  float* out = (float*)d_out;

  int BN = in_sizes[1];              // B*N rows
  int blocks = BN / (WAVES * 16);    // 128 rows per block
  improved_predictor_kernel<<<blocks, THREADS, 0, stream>>>(
      features, last_value, W_ih, W_hh, b_ih, b_hh, Wp, bp,
      Wo1, bo1, Wo2, bo2, Wg1, bg1, Wg2, bg2, log_decay, out);
}
